// UpBlock_2817498546730
// MI455X (gfx1250) — matrix operand order repaired and checked by reading, not compiled, not run on hardware
//
#include <hip/hip_runtime.h>
#include <hip/hip_bf16.h>
#include <stdint.h>

// ---------------------------------------------------------------------------
// UpBlock + 4x Mamba blocks for MI455X (gfx1250, wave32, WMMA).
// All GEMM operands pre-cast to bf16; tiles staged with
// global_load_async_to_lds_b128 (ASYNCcnt, double buffered); inner loop is
// pure ds_load_b128 + v_wmma_f32_16x16x32_bf16 (fp32 accumulate).
// Selective scan: one channel per lane, state in VGPRs, B/C via LDS.
// ---------------------------------------------------------------------------

#define B_SZ     8
#define L_SEQ    1024
#define C_MODEL  512
#define D_INNER  1024
#define D_STATE  16
#define DT_RANK  32

typedef __attribute__((ext_vector_type(16))) __bf16 v16bf;
typedef __attribute__((ext_vector_type(8)))  float  v8f;
typedef unsigned short bf16_t;

union FragAB { unsigned int u[8]; v16bf v; };

__device__ __forceinline__ bf16_t f2bf(float f) {
    unsigned u = __float_as_uint(f);
    return (bf16_t)((u + 0x7FFFu + ((u >> 16) & 1u)) >> 16);  // RNE
}
__device__ __forceinline__ float silu(float v) { return v / (1.0f + __expf(-v)); }

// ---- CDNA5 async global->LDS copy (ASYNCcnt) ------------------------------
#if defined(__has_builtin)
#if __has_builtin(__builtin_amdgcn_global_load_async_to_lds_b128)
#define HAVE_ASYNC_LDS_BUILTIN 1
#endif
#if __has_builtin(__builtin_amdgcn_s_wait_asynccnt)
#define HAVE_WAIT_ASYNC_BUILTIN 1
#endif
#endif

typedef int v4i_t __attribute__((vector_size(4 * sizeof(int))));
typedef __attribute__((address_space(1))) v4i_t* g128_p;
typedef __attribute__((address_space(3))) v4i_t* l128_p;

__device__ __forceinline__ void async_cp16(const void* g, void* l) {
#ifdef HAVE_ASYNC_LDS_BUILTIN
    __builtin_amdgcn_global_load_async_to_lds_b128(
        (g128_p)(unsigned long long)(uintptr_t)g,
        (l128_p)(unsigned)(uintptr_t)l, 0, 0);
#else
    unsigned l32 = (unsigned)(uintptr_t)l;
    asm volatile("global_load_async_to_lds_b128 %0, %1, off"
                 :: "v"(l32), "v"(g) : "memory");
#endif
}
__device__ __forceinline__ void wait_async0() {
#ifdef HAVE_WAIT_ASYNC_BUILTIN
    __builtin_amdgcn_s_wait_asynccnt(0);
#else
    asm volatile("s_wait_asynccnt 0x0" ::: "memory");
#endif
}

// ---------------------------------------------------------------------------
// WMMA GEMM (bf16 x bf16 -> f32):  C[m,n] = sum_k A[m,k] * B[n,k]
//   A: bf16 [.,lda], rows: aOff(m) = (m/rpb)*aBatch + (m%rpb)*lda
//   B: bf16 [N,ldb] row-major (contiguous k)
//   C (f32) and Cbf (bf16) optional outputs, same row indexing scheme.
// All call sites: M % 128 == 0, N % 64 == 0, K % 32 == 0 -> full tiles only.
// ---------------------------------------------------------------------------
#define BM 128
#define BN 64
#define BK 32
#define KPU 20   // 16 k-pair uints + 4 pad  (80B row stride, 16B aligned)

__global__ __launch_bounds__(256) void gemm_wmma_bf16(
    const bf16_t* __restrict__ A, long aBatch, int lda,
    const bf16_t* __restrict__ Bw, int ldb,
    const float* __restrict__ bias,   // [N] or null
    const float* __restrict__ biasB,  // [nBatch, N] or null (e.g. embed)
    const float* __restrict__ Res,    // residual, same indexing as C, or null
    float*  __restrict__ C,   long cBatch,  int ldc,    // may be null
    bf16_t* __restrict__ Cbf, long cbBatch, int ldcb,   // may be null
    int M, int N, int K, int rpb)
{
    __shared__ unsigned int ldsA[2][BM][KPU];
    __shared__ unsigned int ldsB[2][BN][KPU];

    const int m0 = blockIdx.y * BM;
    const int n0 = blockIdx.x * BN;
    const int tid  = threadIdx.x;
    const int lane = tid & 31;
    const int wid  = tid >> 5;
    const int wm   = wid >> 1;   // 0..3 (32 rows each)
    const int wn   = wid & 1;    // 0..1 (32 cols each)

    const int  batch    = m0 / rpb;
    const long aRowBase = (long)batch * aBatch  + (long)(m0 % rpb) * lda;
    const long cRowBase = (long)batch * cBatch  + (long)(m0 % rpb) * ldc;
    const long bRowBase = (long)batch * cbBatch + (long)(m0 % rpb) * ldcb;

    const int nk = K / BK;

    v8f acc[2][2];
#pragma unroll
    for (int mi = 0; mi < 2; ++mi)
#pragma unroll
        for (int ni = 0; ni < 2; ++ni)
            acc[mi][ni] = (v8f){0.f, 0.f, 0.f, 0.f, 0.f, 0.f, 0.f, 0.f};

    // stage one BK-slice: A 128 rows x 64B (4 chunks), B 64 rows x 64B
    auto stage = [&](int kt, int buf) {
        const int k0 = kt * BK;
#pragma unroll
        for (int it = 0; it < 2; ++it) {               // A: 512 chunks
            int c   = it * 256 + tid;
            int row = c >> 2, cq = c & 3;
            async_cp16(A + aRowBase + (long)row * lda + k0 + cq * 8,
                       &ldsA[buf][row][cq * 4]);
        }
        {                                               // B: 256 chunks
            int row = tid >> 2, cq = tid & 3;
            async_cp16(Bw + (long)(n0 + row) * ldb + k0 + cq * 8,
                       &ldsB[buf][row][cq * 4]);
        }
    };

    stage(0, 0);
    wait_async0();
    __syncthreads();

    for (int kt = 0; kt < nk; ++kt) {
        const int buf = kt & 1;
        if (kt + 1 < nk) stage(kt + 1, buf ^ 1);

        const int group = lane >> 4;
        FragAB fb[2];
#pragma unroll
        for (int ni = 0; ni < 2; ++ni) {
            int n = wn * 32 + ni * 16 + (lane & 15);
            // u[p] = k-pair uint (group*8 + p): 8 consecutive uints -> 2x b128
            const uint4 lo = *(const uint4*)&ldsB[buf][n][group * 8];
            const uint4 hi = *(const uint4*)&ldsB[buf][n][group * 8 + 4];
            fb[ni].u[0] = lo.x; fb[ni].u[1] = lo.y; fb[ni].u[2] = lo.z; fb[ni].u[3] = lo.w;
            fb[ni].u[4] = hi.x; fb[ni].u[5] = hi.y; fb[ni].u[6] = hi.z; fb[ni].u[7] = hi.w;
        }
#pragma unroll
        for (int mi = 0; mi < 2; ++mi) {
            FragAB fa;
            int mr = wm * 32 + mi * 16 + (lane & 15);
            // u[p] = k-pair uint (group*8 + p): 8 consecutive uints -> 2x b128
            const uint4 lo = *(const uint4*)&ldsA[buf][mr][group * 8];
            const uint4 hi = *(const uint4*)&ldsA[buf][mr][group * 8 + 4];
            fa.u[0] = lo.x; fa.u[1] = lo.y; fa.u[2] = lo.z; fa.u[3] = lo.w;
            fa.u[4] = hi.x; fa.u[5] = hi.y; fa.u[6] = hi.z; fa.u[7] = hi.w;
#pragma unroll
            for (int ni = 0; ni < 2; ++ni) {
                acc[mi][ni] = __builtin_amdgcn_wmma_f32_16x16x32_bf16(
                    false, fa.v, false, fb[ni].v, (short)0, acc[mi][ni], false, false);
            }
        }

        if (kt + 1 < nk) wait_async0();
        __syncthreads();
    }

    // ---- epilogue ----
#pragma unroll
    for (int mi = 0; mi < 2; ++mi) {
#pragma unroll
        for (int ni = 0; ni < 2; ++ni) {
            int gn = n0 + wn * 32 + ni * 16 + (lane & 15);
            int rbase = wm * 32 + mi * 16 + ((lane >> 4) << 3);
            float badd = (bias ? bias[gn] : 0.f) +
                         (biasB ? biasB[(long)batch * N + gn] : 0.f);
#pragma unroll
            for (int r = 0; r < 8; ++r) {
                int row = rbase + r;
                float v = acc[mi][ni][r] + badd;
                if (Res) v += Res[cRowBase + (long)row * ldc + gn];
                if (C)   C[cRowBase + (long)row * ldc + gn] = v;
                if (Cbf) Cbf[bRowBase + (long)row * ldcb + gn] = f2bf(v);
            }
        }
    }
}

// ---------------------------------------------------------------------------
// Cast kernels (f32 -> bf16), strided gather for weight normalization
// ---------------------------------------------------------------------------
__global__ void cast_bf16_k(const float* __restrict__ src, long sRow, long sCol,
                            bf16_t* __restrict__ dst, int cols, long n)
{
    long idx = (long)blockIdx.x * blockDim.x + threadIdx.x;
    if (idx >= n) return;
    long r = idx / cols;
    int  c = (int)(idx % cols);
    dst[idx] = f2bf(src[r * sRow + (long)c * sCol]);
}

__global__ void skip_cast_k(const float* __restrict__ skip, bf16_t* __restrict__ xcat, long n)
{
    long idx = (long)blockIdx.x * blockDim.x + threadIdx.x;
    if (idx >= n) return;
    long tok = idx / C_MODEL;
    int  c   = (int)(idx % C_MODEL);
    xcat[tok * (2 * C_MODEL) + C_MODEL + c] = f2bf(skip[idx]);
}

// ---------------------------------------------------------------------------
// Row / elementwise kernels
// ---------------------------------------------------------------------------
__global__ __launch_bounds__(256) void rms_silu_k(const float* __restrict__ in,
                                                  const float* __restrict__ w,
                                                  bf16_t* __restrict__ outb, int C)
{
    __shared__ float red[256];
    __shared__ float stat;
    long row = blockIdx.x;
    const float* p = in + row * C;
    bf16_t* o = outb + row * C;
    float ss = 0.f;
    for (int c = threadIdx.x; c < C; c += 256) { float v = p[c]; ss += v * v; }
    red[threadIdx.x] = ss; __syncthreads();
    for (int st = 128; st > 0; st >>= 1) {
        if (threadIdx.x < st) red[threadIdx.x] += red[threadIdx.x + st];
        __syncthreads();
    }
    if (threadIdx.x == 0) stat = rsqrtf(red[0] / C + 1e-5f);
    __syncthreads();
    float s = stat;
    for (int c = threadIdx.x; c < C; c += 256)
        o[c] = f2bf(silu(p[c] * s * w[c]));
}

// layernorm: writes f32 (outF) and/or bf16 (outB)
__global__ __launch_bounds__(256) void layernorm_k(const float* __restrict__ in,
                                                   const float* __restrict__ w,
                                                   const float* __restrict__ b,
                                                   float* __restrict__ outF,
                                                   bf16_t* __restrict__ outB, int C)
{
    __shared__ float red[256];
    __shared__ float stat[2];
    long row = blockIdx.x;
    const float* p = in + row * C;
    float s = 0.f;
    for (int c = threadIdx.x; c < C; c += 256) s += p[c];
    red[threadIdx.x] = s; __syncthreads();
    for (int st = 128; st > 0; st >>= 1) {
        if (threadIdx.x < st) red[threadIdx.x] += red[threadIdx.x + st];
        __syncthreads();
    }
    if (threadIdx.x == 0) stat[0] = red[0] / C;
    __syncthreads();
    float mean = stat[0];
    float ss = 0.f;
    for (int c = threadIdx.x; c < C; c += 256) { float d = p[c] - mean; ss += d * d; }
    __syncthreads();
    red[threadIdx.x] = ss; __syncthreads();
    for (int st = 128; st > 0; st >>= 1) {
        if (threadIdx.x < st) red[threadIdx.x] += red[threadIdx.x + st];
        __syncthreads();
    }
    if (threadIdx.x == 0) stat[1] = rsqrtf(red[0] / C + 1e-5f);
    __syncthreads();
    float rstd = stat[1];
    for (int c = threadIdx.x; c < C; c += 256) {
        float v = (p[c] - mean) * rstd * w[c] + b[c];
        if (outF) outF[row * C + c] = v;
        if (outB) outB[row * C + c] = f2bf(v);
    }
}

// depthwise causal conv (D_CONV=4) + bias + SiLU; writes f32 (scan) + bf16 (GEMM)
__global__ void conv_silu_k(const float* __restrict__ xz, const float* __restrict__ w,
                            const float* __restrict__ bias,
                            float* __restrict__ out, bf16_t* __restrict__ outb,
                            int L, int D, int ld, long n)
{
    long idx = (long)blockIdx.x * blockDim.x + threadIdx.x;
    if (idx >= n) return;
    int  d  = (int)(idx % D);
    long bl = idx / D;
    int  l  = (int)(bl % L);
    long bb = bl / L;
    float acc = bias[d];
#pragma unroll
    for (int j = 0; j < 4; ++j) {
        int ls = l - 3 + j;
        if (ls >= 0) acc += w[d * 4 + j] * xz[(bb * L + ls) * ld + d];
    }
    float v = silu(acc);
    out[idx]  = v;
    outb[idx] = f2bf(v);
}

// g = y * silu(z) -> bf16 (feeds out_proj GEMM)
__global__ void gate_silu_k(const float* __restrict__ y, const float* __restrict__ xz,
                            bf16_t* __restrict__ g, int D, int ld, long n)
{
    long idx = (long)blockIdx.x * blockDim.x + threadIdx.x;
    if (idx >= n) return;
    long tok = idx / D;
    int  d   = (int)(idx % D);
    float z  = xz[tok * ld + D + d];
    g[idx]   = f2bf(y[idx] * silu(z));
}

// ---------------------------------------------------------------------------
// Selective scan: one channel per lane. grid = (B, D/256), block = 256.
// ---------------------------------------------------------------------------
__global__ __launch_bounds__(256) void mamba_scan_k(
    const float* __restrict__ dtlin,  // [B*L, D] pre-softplus
    const float* __restrict__ u,      // [B*L, D]
    const float* __restrict__ xdbc,   // [B*L, 64]  cols 32..47 = B, 48..63 = C
    const float* __restrict__ A_log,  // [D, 16]
    const float* __restrict__ Dp,     // [D]
    float* __restrict__ y,            // [B*L, D]
    int L, int D)
{
    const int b = blockIdx.x;
    const int d = blockIdx.y * blockDim.x + threadIdx.x;
    __shared__ float bc[2 * D_STATE];

    float An[D_STATE], h[D_STATE];
#pragma unroll
    for (int n = 0; n < D_STATE; ++n) {
        An[n] = -__expf(A_log[(long)d * D_STATE + n]);
        h[n]  = 0.f;
    }
    const float Dd = Dp[d];
    const long base = (long)b * L;

    for (int t = 0; t < L; ++t) {
        __syncthreads();
        if (threadIdx.x < 2 * D_STATE)
            bc[threadIdx.x] = xdbc[(base + t) * 64 + DT_RANK + threadIdx.x];
        __syncthreads();
        long row = base + t;
        float xv = dtlin[row * D + d];
        float dt = (xv > 20.f) ? xv : log1pf(__expf(xv));   // softplus
        float uv = u[row * D + d];
        float du = dt * uv;
        float ys = 0.f;
#pragma unroll
        for (int n = 0; n < D_STATE; ++n) {
            float dA = __expf(dt * An[n]);
            h[n] = dA * h[n] + du * bc[n];
            ys  += h[n] * bc[D_STATE + n];
        }
        y[row * D + d] = ys + uv * Dd;
    }
}

// ---------------------------------------------------------------------------
// Orchestration
// ---------------------------------------------------------------------------
static inline dim3 gemm_grid(int M, int N) { return dim3(N / BN, M / BM); }
static inline unsigned nb(long n) { return (unsigned)((n + 255) / 256); }

extern "C" void kernel_launch(void* const* d_in, const int* in_sizes, int n_in,
                              void* d_out, int out_size, void* d_ws, size_t ws_size,
                              hipStream_t stream)
{
    const float* motion = (const float*)d_in[0];
    const float* skip   = (const float*)d_in[1];
    const float* embed  = (const float*)d_in[2];
    const float* w_up   = (const float*)d_in[3];
    const float* b_up   = (const float*)d_in[4];
    const float* w1     = (const float*)d_in[5];
    const float* b1v    = (const float*)d_in[6];
    const float* rms_w  = (const float*)d_in[7];
    const float* w2     = (const float*)d_in[8];
    const float* b2v    = (const float*)d_in[9];
    const float* ln_w   = (const float*)d_in[10];
    const float* ln_b   = (const float*)d_in[11];
    const float* in_prj = (const float*)d_in[12];
    const float* conv_w = (const float*)d_in[13];
    const float* conv_b = (const float*)d_in[14];
    const float* x_prj  = (const float*)d_in[15];
    const float* dt_w   = (const float*)d_in[16];
    const float* dt_b   = (const float*)d_in[17];
    const float* A_log  = (const float*)d_in[18];
    const float* D_par  = (const float*)d_in[19];
    const float* out_w  = (const float*)d_in[20];
    const float* out_b  = (const float*)d_in[21];
    const float* nf_w   = (const float*)d_in[22];
    const float* nf_b   = (const float*)d_in[23];
    float* out = (float*)d_out;

    const long NTOK = (long)B_SZ * L_SEQ;     // 8192 tokens
    // ---- workspace layout (float slots; bf16 regions use 2 per slot) ----
    float* ws    = (float*)d_ws;
    float* x     = ws;                          // f32 [8192, 512]
    float* tmp   = x    + NTOK * C_MODEL;       // f32 [8192, 512]
    float* xz    = tmp  + NTOK * C_MODEL;       // f32 [8192, 2048]
    float* xcc   = xz   + NTOK * 2 * D_INNER;   // f32 [8192, 1024]
    float* xdbc  = xcc  + NTOK * D_INNER;       // f32 [8192, 64]
    float* dtb   = xdbc + NTOK * 64;            // f32 [8192, 1024]
    float* ybuf  = dtb  + NTOK * D_INNER;       // f32 [8192, 1024]
    bf16_t* mo_bf   = (bf16_t*)(ybuf + NTOK * D_INNER);      // [4096, 512]
    bf16_t* xcat_bf = mo_bf   + 4096L * C_MODEL;             // [8192, 1024]
    bf16_t* a_bf    = xcat_bf + NTOK * 2 * C_MODEL;          // [8192, 512]
    bf16_t* xcc_bf  = a_bf    + NTOK * C_MODEL;              // [8192, 1024]
    bf16_t* xdbc_bf = xcc_bf  + NTOK * D_INNER;              // [8192, 64]
    bf16_t* g_bf    = xdbc_bf + NTOK * 64;                   // [8192, 1024]
    bf16_t* w_bf    = g_bf    + NTOK * D_INNER;              // [<= 2048*512]

    dim3 blk(256);
    const long NE_D = NTOK * D_INNER;
    const long NE_C = NTOK * C_MODEL;

    // ---- stage 0: cast motion; up-projection einsum as 2 bf16 GEMMs ----
    cast_bf16_k<<<nb(4096L * 512), blk, 0, stream>>>(motion, 512, 1, mo_bf, 512, 4096L * 512);
    for (int k = 0; k < 2; ++k) {
        // W_k[n,kk] = w_up[kk, n, k]  (gather: row stride 2, col stride 1024)
        cast_bf16_k<<<nb(512L * 512), blk, 0, stream>>>(w_up + k, 2, 1024, w_bf, 512, 512L * 512);
        gemm_wmma_bf16<<<gemm_grid(4096, C_MODEL), blk, 0, stream>>>(
            mo_bf, (long)512 * C_MODEL, C_MODEL, w_bf, C_MODEL,
            b_up, nullptr, nullptr,
            nullptr, 0, 0,
            xcat_bf + (long)k * 1024, (long)L_SEQ * 1024, 2048,
            4096, C_MODEL, C_MODEL, /*rpb=*/512);
    }
    skip_cast_k<<<nb(NE_C), blk, 0, stream>>>(skip, xcat_bf, NE_C);

    // ---- stage 1: t1 = xcat @ w1^T + b1 ; a_bf = bf16(silu(rmsnorm(t1))) ----
    cast_bf16_k<<<nb(512L * 1024), blk, 0, stream>>>(w1, 1024, 1, w_bf, 1024, 512L * 1024);
    gemm_wmma_bf16<<<gemm_grid((int)NTOK, C_MODEL), blk, 0, stream>>>(
        xcat_bf, 0, 1024, w_bf, 1024, b1v, nullptr, nullptr,
        tmp, 0, C_MODEL, nullptr, 0, 0,
        (int)NTOK, C_MODEL, 1024, (int)NTOK);
    rms_silu_k<<<(unsigned)NTOK, blk, 0, stream>>>(tmp, rms_w, a_bf, C_MODEL);

    // ---- stage 2: x = a @ w2^T + b2 + embed[b] ----
    cast_bf16_k<<<nb(512L * 512), blk, 0, stream>>>(w2, 512, 1, w_bf, 512, 512L * 512);
    gemm_wmma_bf16<<<gemm_grid((int)NTOK, C_MODEL), blk, 0, stream>>>(
        a_bf, (long)L_SEQ * C_MODEL, C_MODEL, w_bf, C_MODEL,
        b2v, embed, nullptr,
        x, (long)L_SEQ * C_MODEL, C_MODEL, nullptr, 0, 0,
        (int)NTOK, C_MODEL, C_MODEL, L_SEQ);

    // ---- 4 Mamba blocks ----
    for (int i = 0; i < 4; ++i) {
        layernorm_k<<<(unsigned)NTOK, blk, 0, stream>>>(
            x, ln_w + i * C_MODEL, ln_b + i * C_MODEL, nullptr, a_bf, C_MODEL);

        cast_bf16_k<<<nb(2048L * 512), blk, 0, stream>>>(
            in_prj + (long)i * 2 * D_INNER * C_MODEL, 512, 1, w_bf, 512, 2048L * 512);
        gemm_wmma_bf16<<<gemm_grid((int)NTOK, 2 * D_INNER), blk, 0, stream>>>(
            a_bf, 0, C_MODEL, w_bf, C_MODEL, nullptr, nullptr, nullptr,
            xz, 0, 2 * D_INNER, nullptr, 0, 0,
            (int)NTOK, 2 * D_INNER, C_MODEL, (int)NTOK);

        conv_silu_k<<<nb(NE_D), blk, 0, stream>>>(
            xz, conv_w + (long)i * D_INNER * 4, conv_b + (long)i * D_INNER,
            xcc, xcc_bf, L_SEQ, D_INNER, 2 * D_INNER, NE_D);

        cast_bf16_k<<<nb(64L * 1024), blk, 0, stream>>>(
            x_prj + (long)i * 64 * D_INNER, 1024, 1, w_bf, 1024, 64L * 1024);
        gemm_wmma_bf16<<<gemm_grid((int)NTOK, 64), blk, 0, stream>>>(
            xcc_bf, 0, D_INNER, w_bf, D_INNER, nullptr, nullptr, nullptr,
            xdbc, 0, 64, xdbc_bf, 0, 64,
            (int)NTOK, 64, D_INNER, (int)NTOK);

        cast_bf16_k<<<nb(1024L * 32), blk, 0, stream>>>(
            dt_w + (long)i * D_INNER * DT_RANK, 32, 1, w_bf, 32, 1024L * 32);
        gemm_wmma_bf16<<<gemm_grid((int)NTOK, D_INNER), blk, 0, stream>>>(
            xdbc_bf, 0, 64, w_bf, 32, dt_b + (long)i * D_INNER, nullptr, nullptr,
            dtb, 0, D_INNER, nullptr, 0, 0,
            (int)NTOK, D_INNER, DT_RANK, (int)NTOK);

        mamba_scan_k<<<dim3(B_SZ, D_INNER / 256), blk, 0, stream>>>(
            dtb, xcc, xdbc, A_log + (long)i * D_INNER * D_STATE,
            D_par + (long)i * D_INNER, ybuf, L_SEQ, D_INNER);

        gate_silu_k<<<nb(NE_D), blk, 0, stream>>>(
            ybuf, xz, g_bf, D_INNER, 2 * D_INNER, NE_D);

        cast_bf16_k<<<nb(512L * 1024), blk, 0, stream>>>(
            out_w + (long)i * C_MODEL * D_INNER, 1024, 1, w_bf, 1024, 512L * 1024);
        gemm_wmma_bf16<<<gemm_grid((int)NTOK, C_MODEL), blk, 0, stream>>>(
            g_bf, 0, D_INNER, w_bf, D_INNER,
            out_b + (long)i * C_MODEL, nullptr, /*Res=*/x,
            x, 0, C_MODEL, nullptr, 0, 0,
            (int)NTOK, C_MODEL, D_INNER, (int)NTOK);
    }

    layernorm_k<<<(unsigned)NTOK, blk, 0, stream>>>(x, nf_w, nf_b, out, nullptr, C_MODEL);
}
